// LSTM_65859028517106
// MI455X (gfx1250) — matrix operand order repaired and checked by reading, not compiled, not run on hardware
//
#include <hip/hip_runtime.h>
#include <hip/hip_bf16.h>

typedef __attribute__((ext_vector_type(16))) __bf16   v16bf;
typedef __attribute__((ext_vector_type(8)))  float    v8f;
typedef __attribute__((ext_vector_type(4)))  float    v4f;
typedef __attribute__((ext_vector_type(4)))  unsigned v4u;

#define HID  512
#define G4   2048   // 4*HID gate width
#define BB   128    // batch
#define TT   1024   // time steps
#define IND  128    // input dim

// ---- workspace layout ------------------------------------------------------
// ushort (bf16) region at byte 0:
//   weights in original G-major layout (rows = gate/out index, cols = K),
//   followed by the recurrent h state (double-buffered per layer) in bf16.
#define OFF_WIH1   0u          // 2048*128
#define OFF_WHH1   262144u     // 2048*512
#define OFF_WIHR   1310720u    // 4 * 2048*512
#define OFF_WHHR   5505024u    // 4 * 2048*512
#define OFF_WOUT   9699328u    // 128*512
#define OFF_HBF    9764864u    // 5 layers * 2 bufs * 128*512 bf16
#define H_BUF      (BB * HID)  // 65536 elements per h buffer
#define BF_BYTES   20840448u   // end of ushort region (256-aligned)
// f32 region after that: c state only.
#define C_FLOATS   (5u * H_BUF)        // 327680
// zero-init span: h (bf16) + c (f32) are contiguous; zero as floats.
#define ZERO_BYTE_OFF 19529728u        // = OFF_HBF * 2
#define ZERO_FLOATS   655360u          // (5*2*H_BUF*2 + C_FLOATS*4) / 4

// ---- helpers ---------------------------------------------------------------
__device__ __forceinline__ float sigm_(float x) {
    return 1.0f / (1.0f + __expf(-x));
}
__device__ __forceinline__ float tanh_(float x) {
    x = fminf(fmaxf(x, -15.0f), 15.0f);
    float e = __expf(2.0f * x);
    return (e - 1.0f) / (e + 1.0f);
}

__device__ __forceinline__ v8f wmma_bf16(v16bf a, v16bf b, v8f c) {
    // D = A(16x32 bf16) * B(32x16 bf16) + C(f32)
    return __builtin_amdgcn_wmma_f32_16x16x32_bf16(
        false, a, false, b, (short)0, c, false, false);
}

// A-frag from a row-major fp32 matrix (layer-1 input x only): convert on load.
// Lane (m = lane&15) holds row m; K runs [kh..kh+7] and [kh+16..kh+23],
// kh = 8*(lane>=16)  (ISA 16-bit A layout).
__device__ __forceinline__ v16bf load_a_f32(const float* __restrict__ A, int lda,
                                            int row, int kbase, int lane) {
    const int m  = lane & 15;
    const int kh = (lane >> 4) << 3;
    const float* p = A + (long)(row + m) * lda + (kbase + kh);
    v4f f0 = ((const v4f*)p)[0];
    v4f f1 = ((const v4f*)p)[1];
    v4f f2 = ((const v4f*)(p + 16))[0];
    v4f f3 = ((const v4f*)(p + 16))[1];
    v16bf a;
#pragma unroll
    for (int i = 0; i < 4; ++i) {
        a[i]      = (__bf16)f0[i];
        a[4 + i]  = (__bf16)f1[i];
        a[8 + i]  = (__bf16)f2[i];
        a[12 + i] = (__bf16)f3[i];
    }
    return a;
}

// A-frag from a row-major bf16 matrix (recurrent h state): pure loads,
// two aligned 16B loads per lane, no conversion.
__device__ __forceinline__ v16bf load_a_bf16(const unsigned short* __restrict__ A,
                                             int lda, int row, int kbase, int lane) {
    const int m  = lane & 15;
    const int kh = (lane >> 4) << 3;
    const unsigned short* p = A + (long)(row + m) * lda + (kbase + kh);
    union { v4u q[2]; v16bf v; } u;
    u.q[0] = *(const v4u*)p;          // K kh .. kh+7
    u.q[1] = *(const v4u*)(p + 16);   // K kh+16 .. kh+23
    return u.v;
}

// B-frag for B = W^T where W is stored G-major (ldk = K) in bf16.
// Lane holds column n = lane&15, K [kh..kh+7] and [kh+16..kh+23], kh = kbase + 8*(lane>=16):
// two aligned 16B loads.
__device__ __forceinline__ v16bf load_b_wt(const unsigned short* __restrict__ W,
                                           int ldk, int gbase, int kbase, int lane) {
    const unsigned short* p =
        W + (long)(gbase + (lane & 15)) * ldk + kbase + ((lane >> 4) << 3);
    union { v4u q[2]; v16bf v; } u;
    u.q[0] = *(const v4u*)p;
    u.q[1] = *(const v4u*)(p + 16);
    return u.v;
}

__device__ __forceinline__ unsigned short f2bf_u16(float f) {
    __bf16 b = (__bf16)f;
    return __builtin_bit_cast(unsigned short, b);
}

// ---- prep kernels ----------------------------------------------------------
__global__ __launch_bounds__(256) void zero_f32_k(float* __restrict__ p, int n) {
    int i = blockIdx.x * blockDim.x + threadIdx.x;
    if (i < n) p[i] = 0.0f;
}

__global__ __launch_bounds__(256) void cvt_bf16_k(const float* __restrict__ src,
                                                  unsigned short* __restrict__ dst,
                                                  int n) {
    int i = blockIdx.x * blockDim.x + threadIdx.x;
    if (i >= n) return;
    dst[i] = f2bf_u16(src[i]);
}

// ---- pipelined LSTM step ---------------------------------------------------
// grid = (8, 4, 6), block = 256 (8 wave32).
// slot z in 0..4: layer z at time t = s - z (B=128 x G4=2048 gates).
//   wave computes 16x16 tiles of i,f,g,o for hidden tile j = blockIdx.y*8+wave,
//   batch rows = blockIdx.x*16, then does the cell pointwise update in-register
//   and stores h(t) as bf16, c(t) as f32.
// slot z == 5: output projection of h5(t = s-5) -> d_out.
__global__ __launch_bounds__(256)
void lstm_step(const float* __restrict__ x,
               const float* __restrict__ bih1, const float* __restrict__ bhh1,
               const float* __restrict__ bih_r, const float* __restrict__ bhh_r,
               const float* __restrict__ bout,
               const unsigned short* __restrict__ wbf,  // weights (bf16)
               unsigned short* __restrict__ hws,        // h state (bf16)
               float* __restrict__ cws,                 // c state (f32)
               float* __restrict__ out, int s) {
    const int lane = threadIdx.x & 31;
    const int wave = threadIdx.x >> 5;
    const int slot = blockIdx.z;
    const int rowb = blockIdx.x << 4;
    const int mh   = (lane >> 4) << 3;   // +8 row offset for upper half-wave
    const int n    = lane & 15;

    if (slot == 5) {  // ---- output projection: out[b][t][:] = h5 @ Wout^T + bout
        int t = s - 5;
        if (t < 0 || t >= TT || blockIdx.y != 0) return;
        const unsigned short* H5 = hws + (8 + (t & 1)) * H_BUF;
        const unsigned short* Wo = wbf + OFF_WOUT;  // 128 x 512 (G-major)
        const int ct = wave;                        // 0..7 output col tile
        v8f acc = {0.f, 0.f, 0.f, 0.f, 0.f, 0.f, 0.f, 0.f};
#pragma unroll 2
        for (int kb = 0; kb < HID; kb += 32) {
            v16bf a = load_a_bf16(H5, HID, rowb, kb, lane);
            v16bf b = load_b_wt(Wo, HID, ct << 4, kb, lane);
            acc = wmma_bf16(a, b, acc);
        }
        const int col = (ct << 4) + n;
        const float bo = bout[col];
#pragma unroll
        for (int r = 0; r < 8; ++r) {
            int m = rowb + r + mh;
            out[((long)m * TT + t) * IND + col] = acc[r] + bo;
        }
        return;
    }

    const int l = slot;
    const int t = s - l;
    if (t < 0 || t >= TT) return;

    const unsigned short *Wih, *Whh;
    const float *bi_, *bh_;
    const unsigned short* Ain_h = nullptr;   // bf16 input (layers >= 1)
    if (l == 0) {
        Wih = wbf + OFF_WIH1;  Whh = wbf + OFF_WHH1;
        bi_ = bih1;            bh_ = bhh1;
    } else {
        Ain_h = hws + ((l - 1) * 2 + (t & 1)) * H_BUF;
        Wih = wbf + OFF_WIHR + (unsigned)(l - 1) * (G4 * HID);
        Whh = wbf + OFF_WHHR + (unsigned)(l - 1) * (G4 * HID);
        bi_ = bih_r + (l - 1) * G4;
        bh_ = bhh_r + (l - 1) * G4;
    }
    const unsigned short* Hprev = hws + (l * 2 + ((t + 1) & 1)) * H_BUF;  // h(t-1)
    unsigned short*       Hout  = hws + (l * 2 + (t & 1)) * H_BUF;        // h(t)
    float*                C     = cws + l * H_BUF;

    const int j  = (blockIdx.y << 3) + wave;  // hidden tile 0..31
    const int g0 = j << 4;

    v8f ai = {0.f,0.f,0.f,0.f,0.f,0.f,0.f,0.f};
    v8f af = {0.f,0.f,0.f,0.f,0.f,0.f,0.f,0.f};
    v8f ag = {0.f,0.f,0.f,0.f,0.f,0.f,0.f,0.f};
    v8f ao = {0.f,0.f,0.f,0.f,0.f,0.f,0.f,0.f};

    // G += Xin @ Wih^T   (one A-frag reused by the 4 gate B-frags)
    if (l == 0) {
#pragma unroll 2
        for (int kb = 0; kb < IND; kb += 32) {
            v16bf a = load_a_f32(x + (long)t * IND, TT * IND, rowb, kb, lane);
            ai = wmma_bf16(a, load_b_wt(Wih, IND, g0,        kb, lane), ai);
            af = wmma_bf16(a, load_b_wt(Wih, IND, g0 +  512, kb, lane), af);
            ag = wmma_bf16(a, load_b_wt(Wih, IND, g0 + 1024, kb, lane), ag);
            ao = wmma_bf16(a, load_b_wt(Wih, IND, g0 + 1536, kb, lane), ao);
        }
    } else {
#pragma unroll 2
        for (int kb = 0; kb < HID; kb += 32) {
            v16bf a = load_a_bf16(Ain_h, HID, rowb, kb, lane);
            ai = wmma_bf16(a, load_b_wt(Wih, HID, g0,        kb, lane), ai);
            af = wmma_bf16(a, load_b_wt(Wih, HID, g0 +  512, kb, lane), af);
            ag = wmma_bf16(a, load_b_wt(Wih, HID, g0 + 1024, kb, lane), ag);
            ao = wmma_bf16(a, load_b_wt(Wih, HID, g0 + 1536, kb, lane), ao);
        }
    }
    // G += Hprev @ Whh^T
#pragma unroll 2
    for (int kb = 0; kb < HID; kb += 32) {
        v16bf a = load_a_bf16(Hprev, HID, rowb, kb, lane);
        ai = wmma_bf16(a, load_b_wt(Whh, HID, g0,        kb, lane), ai);
        af = wmma_bf16(a, load_b_wt(Whh, HID, g0 +  512, kb, lane), af);
        ag = wmma_bf16(a, load_b_wt(Whh, HID, g0 + 1024, kb, lane), ag);
        ao = wmma_bf16(a, load_b_wt(Whh, HID, g0 + 1536, kb, lane), ao);
    }

    const int hc = g0 + n;
    const float bI = bi_[hc]        + bh_[hc];
    const float bF = bi_[512 + hc]  + bh_[512 + hc];
    const float bG = bi_[1024 + hc] + bh_[1024 + hc];
    const float bO = bi_[1536 + hc] + bh_[1536 + hc];

#pragma unroll
    for (int r = 0; r < 8; ++r) {
        const int  m   = rowb + r + mh;
        const long idx = (long)m * HID + hc;
        const float gi = ai[r] + bI;
        const float gf = af[r] + bF;
        const float gg = ag[r] + bG;
        const float go = ao[r] + bO;
        const float c2 = sigm_(gf) * C[idx] + sigm_(gi) * tanh_(gg);
        const float h2 = sigm_(go) * tanh_(c2);
        C[idx]    = c2;
        Hout[idx] = f2bf_u16(h2);
    }
}

// ---- host ------------------------------------------------------------------
extern "C" void kernel_launch(void* const* d_in, const int* in_sizes, int n_in,
                              void* d_out, int out_size, void* d_ws, size_t ws_size,
                              hipStream_t stream) {
    (void)in_sizes; (void)n_in; (void)out_size; (void)ws_size;
    const float* x     = (const float*)d_in[0];
    const float* Wih1  = (const float*)d_in[1];
    const float* Whh1  = (const float*)d_in[2];
    const float* bih1  = (const float*)d_in[3];
    const float* bhh1  = (const float*)d_in[4];
    const float* Wih_r = (const float*)d_in[5];
    const float* Whh_r = (const float*)d_in[6];
    const float* bih_r = (const float*)d_in[7];
    const float* bhh_r = (const float*)d_in[8];
    const float* Wout  = (const float*)d_in[9];
    const float* bout  = (const float*)d_in[10];

    unsigned short* wbf = (unsigned short*)d_ws;
    unsigned short* hws = wbf + OFF_HBF;
    float*          cws = (float*)((char*)d_ws + BF_BYTES);

    // Prep (runs every call; deterministic): zero h/c state, weights -> bf16.
    zero_f32_k<<<(ZERO_FLOATS + 255) / 256, 256, 0, stream>>>(
        (float*)((char*)d_ws + ZERO_BYTE_OFF), (int)ZERO_FLOATS);
    cvt_bf16_k<<<(G4 * IND + 255) / 256, 256, 0, stream>>>(Wih1,  wbf + OFF_WIH1, G4 * IND);
    cvt_bf16_k<<<(G4 * HID + 255) / 256, 256, 0, stream>>>(Whh1,  wbf + OFF_WHH1, G4 * HID);
    cvt_bf16_k<<<(4 * G4 * HID + 255) / 256, 256, 0, stream>>>(Wih_r, wbf + OFF_WIHR, 4 * G4 * HID);
    cvt_bf16_k<<<(4 * G4 * HID + 255) / 256, 256, 0, stream>>>(Whh_r, wbf + OFF_WHHR, 4 * G4 * HID);
    cvt_bf16_k<<<(IND * HID + 255) / 256, 256, 0, stream>>>(Wout,  wbf + OFF_WOUT, IND * HID);

    // Wavefront-pipelined recurrence: 1024 + 5 pipeline steps.
    dim3 grid(8, 4, 6), block(256);
    for (int s = 0; s < TT + 5; ++s) {
        lstm_step<<<grid, block, 0, stream>>>(x, bih1, bhh1, bih_r, bhh_r, bout,
                                              wbf, hws, cws, (float*)d_out, s);
    }
}
